// RoPEAttention_wMask_47983374631536
// MI455X (gfx1250) — matrix operand order repaired and checked by reading, not compiled, not run on hardware
//
#include <hip/hip_runtime.h>
#include <hip/hip_bf16.h>

// ---------------------------------------------------------------------------
// RoPE attention for MI455X (gfx1250), wave32 + WMMA f16 16x16x32, f32 accum.
// B=2, M=2048, D=1024, H=16, hd=64. ~95 GFLOP vs ~50MB traffic => compute
// bound; all GEMMs + attention run on v_wmma_f32_16x16x32_f16, data movement
// through the gfx1250 async global->LDS copy engine (ASYNCcnt) where available.
// ---------------------------------------------------------------------------

typedef __attribute__((ext_vector_type(16))) _Float16 v16h;
typedef __attribute__((ext_vector_type(8)))  _Float16 v8h;
typedef __attribute__((ext_vector_type(8)))  float    v8f;
typedef int v4i __attribute__((vector_size(4 * sizeof(int))));

#define BB    2
#define MSEQ  2048
#define DD    1024
#define HH    16
#define HD    64
#define MTOT  (BB * MSEQ)

#if defined(__has_builtin)
#  if __has_builtin(__builtin_amdgcn_global_load_async_to_lds_b128)
#    define HAVE_ASYNC_LDS 1
#  endif
#  if __has_builtin(__builtin_amdgcn_s_wait_asynccnt)
#    define HAVE_WAIT_ASYNC 1
#  endif
#endif
#ifndef HAVE_ASYNC_LDS
#  define HAVE_ASYNC_LDS 0
#endif
#ifndef HAVE_WAIT_ASYNC
#  define HAVE_WAIT_ASYNC 0
#endif

// async 16B global->LDS copy (ASYNCcnt path); generic fallback otherwise.
__device__ inline void async_cp16(const _Float16* g, _Float16* l) {
#if HAVE_ASYNC_LDS
    __builtin_amdgcn_global_load_async_to_lds_b128(
        (__attribute__((address_space(1))) v4i*)(g),
        (__attribute__((address_space(3))) v4i*)(l), 0, 0);
#else
    *(v8h*)l = *(const v8h*)g;
#endif
}
__device__ inline void async_wait0() {
#if HAVE_WAIT_ASYNC
    __builtin_amdgcn_s_wait_asynccnt(0);
#elif HAVE_ASYNC_LDS
    asm volatile("s_wait_asynccnt 0" ::: "memory");
#endif
}

// ---- WMMA fragment loaders (contiguous k order, same for A and B) ------

// A 16x32 f16: lane row = lane&15; e -> K=16*(l>=16)+e
__device__ inline v16h load_a_frag(const _Float16* A, int lda, int row0, int k0, int lane) {
    int r  = row0 + (lane & 15);
    int kh = (lane >> 4) << 4;
    const _Float16* p = A + (long)r * lda + k0 + kh;
    v8h lo = *(const v8h*)(p);
    v8h hi = *(const v8h*)(p + 8);
    v16h o;
#pragma unroll
    for (int e = 0; e < 8; ++e) { o[e] = lo[e]; o[e + 8] = hi[e]; }
    return o;
}

// B 32x16 f16 with B[k][n] = Wt[n][k]: lane col N = lane&15; e -> K=16*(l>=16)+e
__device__ inline v16h load_bt_frag(const _Float16* Wt, int ldw, int n0, int k0, int lane) {
    int n  = n0 + (lane & 15);
    int kh = (lane >> 4) << 4;
    return *(const v16h*)(Wt + (long)n * ldw + k0 + kh);
}

// same as load_bt_frag but row-clamped (causal diagonal block reads past M)
__device__ inline v16h load_k_frag(const _Float16* Kb, int n0, int k0, int lane) {
    int n = n0 + (lane & 15);
    n = n < MSEQ ? n : MSEQ - 1;
    int kh = (lane >> 4) << 4;
    return *(const v16h*)(Kb + (long)n * HD + k0 + kh);
}

// LDS-tile fragment loads (row stride 40 f16 = 80B to spread banks)
__device__ inline v16h lds_a_frag(const _Float16* base, int row0, int lane) {
    int r  = row0 + (lane & 15);
    int kh = (lane >> 4) << 4;
    const _Float16* p = base + r * 40 + kh;
    v8h lo = *(const v8h*)(p);
    v8h hi = *(const v8h*)(p + 8);
    v16h o;
#pragma unroll
    for (int e = 0; e < 8; ++e) { o[e] = lo[e]; o[e + 8] = hi[e]; }
    return o;
}
__device__ inline v16h lds_bt_frag(const _Float16* base, int n0, int lane) {
    int n  = n0 + (lane & 15);
    int kh = (lane >> 4) << 4;
    const _Float16* p = base + n * 40 + kh;
    v8h lo = *(const v8h*)(p);
    v8h hi = *(const v8h*)(p + 8);
    v16h o;
#pragma unroll
    for (int e = 0; e < 8; ++e) { o[e] = lo[e]; o[e + 8] = hi[e]; }
    return o;
}

__device__ inline v8f wmma_f16(v16h a, v16h b, v8f c) {
    return __builtin_amdgcn_wmma_f32_16x16x32_f16(false, a, false, b, (short)0, c, false, false);
}

// ---- fp32 -> fp16 convert ---------------------------------------------------
__global__ void cvt_f32_f16(const float* __restrict__ in, _Float16* __restrict__ out, int n) {
    int i = blockIdx.x * blockDim.x + threadIdx.x;
    if (i < n) out[i] = (_Float16)in[i];
}

// ---- GEMM: C[MTOT,DD] = A[MTOT,DD] @ Wt[DD,DD]^T, double-buffered async LDS
// EPI 0: RoPE epilogue -> f16 [MTOT,DD]; EPI 1: +bias -> f16 [B,H,M,HD];
// EPI 2: +bias -> f32 [MTOT,DD].  256 thr = 8 waves (4Mx2N), wave tile 32x32.
template <int EPI>
__global__ __launch_bounds__(256) void gemm_wmma(const _Float16* __restrict__ A,
                                                 const _Float16* __restrict__ Wt,
                                                 const float* __restrict__ bias,
                                                 void* __restrict__ Cout) {
    __shared__ __align__(32) _Float16 Abuf[2][128 * 40];  // 128x32 tile, pad 40
    __shared__ __align__(32) _Float16 Bbuf[2][64 * 40];   // 64x32 tile, pad 40

    const int t      = threadIdx.x;
    const int lane   = t & 31;
    const int wave   = t >> 5;
    const int rowblk = blockIdx.y * 128;
    const int colblk = blockIdx.x * 64;
    const int row0   = (wave >> 1) * 32;   // local in block tile
    const int col0   = (wave & 1) * 32;

    auto stage = [&](int buf, int k0) {
#pragma unroll
        for (int i = 0; i < 4; ++i) {            // A: 1024 x 16B chunks / 256 thr
            int c = i * 256 + t;
            int r = c >> 2, s = (c & 3) * 8;
            async_cp16(A + (long)(rowblk + r) * DD + k0 + s, &Abuf[buf][r * 40 + s]);
        }
        {                                        // B: 256 x 16B chunks
            int r = t >> 2, s = (t & 3) * 8;
            async_cp16(Wt + (long)(colblk + r) * DD + k0 + s, &Bbuf[buf][r * 40 + s]);
        }
    };

    v8f acc[2][2] = {};
    stage(0, 0);
    async_wait0();
    __syncthreads();

    for (int k0 = 0; k0 < DD; k0 += 32) {
        const int buf = (k0 >> 5) & 1;
        if (k0 + 32 < DD) stage(buf ^ 1, k0 + 32);   // overlap copy with WMMA
        v16h a0 = lds_a_frag(&Abuf[buf][0], row0,      lane);
        v16h a1 = lds_a_frag(&Abuf[buf][0], row0 + 16, lane);
        v16h b0 = lds_bt_frag(&Bbuf[buf][0], col0,      lane);
        v16h b1 = lds_bt_frag(&Bbuf[buf][0], col0 + 16, lane);
        acc[0][0] = wmma_f16(a0, b0, acc[0][0]);
        acc[0][1] = wmma_f16(a0, b1, acc[0][1]);
        acc[1][0] = wmma_f16(a1, b0, acc[1][0]);
        acc[1][1] = wmma_f16(a1, b1, acc[1][1]);
        async_wait0();
        __syncthreads();
    }

    const int half = lane >> 4;
    const int nl   = lane & 15;
#pragma unroll
    for (int i = 0; i < 2; ++i) {
#pragma unroll
        for (int j = 0; j < 2; ++j) {
            int col = colblk + col0 + 16 * j + nl;
#pragma unroll
            for (int e = 0; e < 8; ++e) {
                int   row = rowblk + row0 + 16 * i + 8 * half + e;
                float v   = acc[i][j][e];
                if (EPI == 0) {
                    // RoPE: pairs are adjacent columns == adjacent lanes (l^1 in-half)
                    float p    = __shfl_xor(v, 1, 32);
                    int   mpos = row & (MSEQ - 1);
                    int   ip   = col >> 1;
                    float theta = __expf(-2.0f * ((float)ip - 1.0f) *
                                         (9.210340371976184f / (float)DD));
                    float ang = (float)mpos * theta;
                    float s, c;
                    __sincosf(ang, &s, &c);
                    float r = (col & 1) ? (v * c - p * s) : (v * c + p * s);
                    ((_Float16*)Cout)[(long)row * DD + col] = (_Float16)r;
                } else if (EPI == 1) {
                    float r  = v + bias[col];
                    int   b  = row >> 11;
                    int   m  = row & (MSEQ - 1);
                    int   h  = col >> 6;
                    int   dd = col & (HD - 1);
                    ((_Float16*)Cout)[(((long)(b * HH + h)) * MSEQ + m) * HD + dd] = (_Float16)r;
                } else {
                    ((float*)Cout)[(long)row * DD + col] = v + bias[col];
                }
            }
        }
    }
}

// ---- Flash attention: per-wave 16 q-rows, 32-key blocks, online softmax ----
// Q,K,V: [B*H, MSEQ, HD] f16. V tile async-staged per wave; P transposed via LDS.
__global__ __launch_bounds__(256) void attn_kernel(const _Float16* __restrict__ Q,
                                                   const _Float16* __restrict__ K,
                                                   const _Float16* __restrict__ V,
                                                   _Float16* __restrict__ O) {
    __shared__ __align__(32) _Float16 pls[8][16 * 32];   // per-wave P tile (1KB)
#if HAVE_ASYNC_LDS
    __shared__ __align__(32) _Float16 vls[8][32 * 64];   // per-wave V tile (4KB)
#endif

    const int lane  = threadIdx.x & 31;
    const int wave  = threadIdx.x >> 5;
    const int bh    = blockIdx.y;
    const int qbase = blockIdx.x * 128 + wave * 16;
    const long base = (long)bh * MSEQ * HD;
    const _Float16* Qb = Q + base;
    const _Float16* Kb = K + base;
    const _Float16* Vb = V + base;

    const int half = lane >> 4;
    const int nl   = lane & 15;

    v16h qf[2];
#pragma unroll
    for (int c = 0; c < 2; ++c) {
        qf[c] = load_a_frag(Qb, HD, qbase, 32 * c, lane);
#pragma unroll
        for (int e = 0; e < 16; ++e) qf[c][e] = qf[c][e] * (_Float16)0.125f;  // 1/sqrt(64)
    }

    v8f   of[4] = {};
    float mrow[8], lrow[8];
#pragma unroll
    for (int e = 0; e < 8; ++e) { mrow[e] = -3.0e38f; lrow[e] = 0.0f; }

    _Float16* myp = &pls[wave][0];

    for (int kb = 0; kb < qbase + 16; kb += 32) {
#if HAVE_ASYNC_LDS
        // stage V[kb..kb+31][0..63] into per-wave LDS (overlaps score WMMAs)
#pragma unroll
        for (int j = 0; j < 8; ++j) {
            int c = j * 32 + lane;
            int r = c >> 3, s = (c & 7) * 8;
            int rg = kb + r; rg = rg < MSEQ ? rg : MSEQ - 1;
            async_cp16(Vb + (long)rg * HD + s, &vls[wave][r * 64 + s]);
        }
#endif
        v8f s0 = {}, s1 = {};
#pragma unroll
        for (int c = 0; c < 2; ++c) {
            v16h k0f = load_k_frag(Kb, kb,      32 * c, lane);
            v16h k1f = load_k_frag(Kb, kb + 16, 32 * c, lane);
            s0 = wmma_f16(qf[c], k0f, s0);
            s1 = wmma_f16(qf[c], k1f, s1);
        }

        const int col0 = kb + nl;
        const int col1 = col0 + 16;
#pragma unroll
        for (int e = 0; e < 8; ++e) {
            int   row = qbase + 8 * half + e;
            float x0  = (col0 <= row) ? s0[e] : -3.0e38f;   // causal mask
            float x1  = (col1 <= row) ? s1[e] : -3.0e38f;
            float mx  = fmaxf(x0, x1);
#pragma unroll
            for (int d = 1; d < 16; d <<= 1) mx = fmaxf(mx, __shfl_xor(mx, d, 32));
            float mnew  = fmaxf(mrow[e], mx);
            float alpha = __expf(mrow[e] - mnew);
            float p0    = __expf(x0 - mnew);
            float p1    = __expf(x1 - mnew);
            float ps    = p0 + p1;
#pragma unroll
            for (int d = 1; d < 16; d <<= 1) ps += __shfl_xor(ps, d, 32);
            lrow[e] = lrow[e] * alpha + ps;
            mrow[e] = mnew;
#pragma unroll
            for (int n = 0; n < 4; ++n) of[n][e] *= alpha;
            myp[(8 * half + e) * 32 + nl]      = (_Float16)p0;
            myp[(8 * half + e) * 32 + nl + 16] = (_Float16)p1;
        }
        async_wait0();            // per-wave: V tile landed
        __threadfence_block();    // same-wave DS ordering for P tile (no barrier)

        v16h pf = load_a_frag(myp, 32, 0, 0, lane);   // P as 16x32 A fragment

#pragma unroll
        for (int n = 0; n < 4; ++n) {
            v16h vf;
#if HAVE_ASYNC_LDS
            const _Float16* vp = &vls[wave][(16 * half) * 64 + 16 * n + nl];
#pragma unroll
            for (int e = 0; e < 16; ++e) vf[e] = vp[e * 64];
#else
#pragma unroll
            for (int e = 0; e < 16; ++e) {
                int rg = kb + 16 * half + e; rg = rg < MSEQ ? rg : MSEQ - 1;
                vf[e] = Vb[(long)rg * HD + 16 * n + nl];
            }
#endif
            of[n] = wmma_f16(pf, vf, of[n]);
        }
        __threadfence_block();
    }

    const int b = bh >> 4, h = bh & (HH - 1);
#pragma unroll
    for (int e = 0; e < 8; ++e) {
        int   m   = qbase + 8 * half + e;
        float inv = 1.0f / lrow[e];
#pragma unroll
        for (int n = 0; n < 4; ++n) {
            int d = h * HD + 16 * n + nl;
            O[((long)(b * MSEQ + m)) * DD + d] = (_Float16)(of[n][e] * inv);
        }
    }
}

// ---------------------------------------------------------------------------
extern "C" void kernel_launch(void* const* d_in, const int* in_sizes, int n_in,
                              void* d_out, int out_size, void* d_ws, size_t ws_size,
                              hipStream_t stream) {
    (void)in_sizes; (void)n_in; (void)out_size; (void)ws_size;
    const float* x   = (const float*)d_in[0];
    const float* wq  = (const float*)d_in[1];
    const float* wk  = (const float*)d_in[2];
    const float* wv  = (const float*)d_in[3];
    const float* ipw = (const float*)d_in[4];
    const float* ipb = (const float*)d_in[5];
    const float* ow  = (const float*)d_in[6];
    const float* ob  = (const float*)d_in[7];

    char*  ws  = (char*)d_ws;
    size_t off = 0;
    auto alloc = [&](size_t bytes) -> _Float16* {
        _Float16* p = (_Float16*)(ws + off);
        off += (bytes + 255) & ~(size_t)255;
        return p;
    };
    _Float16* xh    = alloc((size_t)MTOT * DD * 2);
    _Float16* wqh   = alloc((size_t)DD * DD * 2);
    _Float16* wkh   = alloc((size_t)DD * DD * 2);
    _Float16* wvh   = alloc((size_t)DD * DD * 2);
    _Float16* iph   = alloc((size_t)3 * DD * DD * 2);
    _Float16* owh   = alloc((size_t)DD * DD * 2);
    _Float16* trope = alloc((size_t)MTOT * DD * 2);
    _Float16* qh2   = alloc((size_t)MTOT * DD * 2);   // [B,H,M,64]
    _Float16* kh2   = alloc((size_t)MTOT * DD * 2);
    _Float16* vh2   = alloc((size_t)MTOT * DD * 2);
    _Float16* oo    = alloc((size_t)MTOT * DD * 2);

    const int NX = MTOT * DD, NW = DD * DD, NP = 3 * DD * DD;
    cvt_f32_f16<<<(NX + 255) / 256, 256, 0, stream>>>(x, xh, NX);
    cvt_f32_f16<<<(NW + 255) / 256, 256, 0, stream>>>(wq, wqh, NW);
    cvt_f32_f16<<<(NW + 255) / 256, 256, 0, stream>>>(wk, wkh, NW);
    cvt_f32_f16<<<(NW + 255) / 256, 256, 0, stream>>>(wv, wvh, NW);
    cvt_f32_f16<<<(NP + 255) / 256, 256, 0, stream>>>(ipw, iph, NP);
    cvt_f32_f16<<<(NW + 255) / 256, 256, 0, stream>>>(ow, owh, NW);

    dim3 grid(DD / 64, MTOT / 128);
    const _Float16* wh3[3] = {wqh, wkh, wvh};
    _Float16*       hb3[3] = {qh2, kh2, vh2};
    for (int i = 0; i < 3; ++i) {
        gemm_wmma<0><<<grid, 256, 0, stream>>>(xh, wh3[i], nullptr, trope);
        gemm_wmma<1><<<grid, 256, 0, stream>>>(trope, iph + (size_t)i * DD * DD,
                                               ipb + i * DD, hb3[i]);
    }
    attn_kernel<<<dim3(MSEQ / 128, BB * HH), 256, 0, stream>>>(qh2, kh2, vh2, oo);
    gemm_wmma<2><<<grid, 256, 0, stream>>>(oo, owh, ob, d_out);
}
